// RobertaLocalAttention_6167573037275
// MI455X (gfx1250) — matrix operand order repaired and checked by reading, not compiled, not run on hardware
//
#include <hip/hip_runtime.h>
#include <cstdint>

// ---------------- problem constants ----------------
#define BB   4
#define SS   4096
#define EE   1024
#define HH   16
#define HD   64
#define NWIN 32
#define WSZ  128
#define BHN  (BB*HH)   // 64
#define N3E  (3*EE)    // 3072

// workspace layout (bytes):
//   xb  bf16[16777216]  @ 0          (32 MiB)
//   wT  bf16[3145728]   @ 33554432   ( 6 MiB)  -- w_qkv transposed, N-major
//   qws bf16[64*4096*64]@ 39845888   (32 MiB)  -- [bh][s][d]
//   kws bf16[...]       @ 73400320   (32 MiB)  -- [bh][s][d]
//   vws bf16[...]       @ 106954752  (32 MiB)  -- [bh][d][s] (transposed)
// total 140,509,184 bytes

typedef __attribute__((ext_vector_type(16))) __bf16 v16bf;
typedef __attribute__((ext_vector_type(8)))  __bf16 v8bf;
typedef __attribute__((ext_vector_type(4)))  __bf16 v4bf;
typedef __attribute__((ext_vector_type(8)))  float  v8f;

union Frag { v16bf v; v8bf h[2]; };

#ifndef __has_builtin
#define __has_builtin(x) 0
#endif

// ---- CDNA5 async copy global->LDS (guarded; fallback keeps compile green) ----
#if __has_builtin(__builtin_amdgcn_global_load_async_to_lds_b128)
#define HAVE_ASYNC 1
// Builtin expects GCC-style int __vector(4) pointers in AS1 (global) / AS3 (LDS).
typedef int v4i_g __attribute__((vector_size(16)));
typedef __attribute__((address_space(1))) v4i_g ga_v4i;
typedef __attribute__((address_space(3))) v4i_g lo_v4i;
__device__ __forceinline__ ga_v4i* as_g(const void* p) {
  return (ga_v4i*)(unsigned long long)(uintptr_t)p;
}
__device__ __forceinline__ lo_v4i* as_l(void* p) {
  // generic->LDS: low 32 bits of a generic LDS pointer are the LDS address
  return (lo_v4i*)(unsigned int)(uintptr_t)p;
}
#else
#define HAVE_ASYNC 0
#endif

__device__ __forceinline__ void async_wait0() {
#if HAVE_ASYNC
#if __has_builtin(__builtin_amdgcn_s_wait_asynccnt)
  __builtin_amdgcn_s_wait_asynccnt(0);
#else
  asm volatile("s_wait_asynccnt 0" ::: "memory");
#endif
#endif
}

#define WMMA_BF16(Av, Bv, Cv) \
  __builtin_amdgcn_wmma_f32_16x16x32_bf16(false, (Av), false, (Bv), (short)0, (Cv), false, false)

// ---------------- pre-pass: fp32 -> bf16 ----------------
__global__ __launch_bounds__(256) void cvt_x_kernel(const float* __restrict__ x,
                                                    __bf16* __restrict__ xb) {
  const size_t i = ((size_t)blockIdx.x * 256 + threadIdx.x) * 4;
  const float4 v = *(const float4*)(x + i);
  v4bf o;
  o[0] = (__bf16)v.x; o[1] = (__bf16)v.y; o[2] = (__bf16)v.z; o[3] = (__bf16)v.w;
  *(v4bf*)(xb + i) = o;
}

// w_qkv [1024][3072] -> wT [3072][1024] (N-major, bf16)
__global__ __launch_bounds__(256) void cvt_wT_kernel(const float* __restrict__ w,
                                                     __bf16* __restrict__ wT) {
  const int idx = blockIdx.x * 256 + threadIdx.x;   // 3,145,728 total
  const int n = idx >> 10, k = idx & 1023;
  wT[idx] = (__bf16)w[(size_t)k * N3E + n];
}

// ---------------- QKV GEMM: [16384x1024] x [1024x3072] ----------------
#define APAD 40   // padded K-stride (elements): 80B rows -> conflict-free b128 reads

__global__ __launch_bounds__(256) void qkv_gemm(
    const __bf16* __restrict__ Axb, const __bf16* __restrict__ BwT,
    const float* __restrict__ bias,
    __bf16* __restrict__ qws, __bf16* __restrict__ kws, __bf16* __restrict__ vws)
{
  __shared__ __bf16 As[2][128 * APAD];   // [m][k]
  __shared__ __bf16 Bs[2][128 * APAD];   // [n][k]
  const int tid = threadIdx.x;
  const int wv = tid >> 5, lane = tid & 31;
  const int ln = lane & 15, g = lane >> 4;
  const int m0 = blockIdx.y * 128, n0 = blockIdx.x * 128;
  const int wm = (wv & 3) * 32, wn = (wv >> 2) * 64;

  v8f acc[2][4];
  const v8f vzero = {0.f,0.f,0.f,0.f,0.f,0.f,0.f,0.f};
#pragma unroll
  for (int i = 0; i < 2; ++i)
#pragma unroll
    for (int j = 0; j < 4; ++j) acc[i][j] = vzero;

  auto stage = [&](int buf, int k0) {
#pragma unroll
    for (int j = 0; j < 2; ++j) {
      const int c = tid + j * 256;            // 512 x 16B chunks per tile
      const int r = c >> 2, kc = (c & 3) * 8;
      const __bf16* gA = Axb + (size_t)(m0 + r) * EE + k0 + kc;
      const __bf16* gB = BwT + (size_t)(n0 + r) * EE + k0 + kc;
      __bf16* lA = &As[buf][r * APAD + kc];
      __bf16* lB = &Bs[buf][r * APAD + kc];
#if HAVE_ASYNC
      __builtin_amdgcn_global_load_async_to_lds_b128(as_g(gA), as_l(lA), 0, 0);
      __builtin_amdgcn_global_load_async_to_lds_b128(as_g(gB), as_l(lB), 0, 0);
#else
      *(v8bf*)lA = *(const v8bf*)gA;
      *(v8bf*)lB = *(const v8bf*)gB;
#endif
    }
  };

  stage(0, 0);
  async_wait0();
  __syncthreads();

  for (int kt = 0; kt < EE / 32; ++kt) {
    const int buf = kt & 1;
    if (kt + 1 < EE / 32) stage(buf ^ 1, (kt + 1) * 32);

    Frag a[2], b[4];
#pragma unroll
    for (int im = 0; im < 2; ++im) {
      const int rm = wm + im * 16 + ln;
      a[im].h[0] = *(const v8bf*)&As[buf][rm * APAD + g * 8];
      a[im].h[1] = *(const v8bf*)&As[buf][rm * APAD + 16 + g * 8];
    }
#pragma unroll
    for (int jn = 0; jn < 4; ++jn) {
      const int rn = wn + jn * 16 + ln;
      b[jn].h[0] = *(const v8bf*)&Bs[buf][rn * APAD + g * 8];
      b[jn].h[1] = *(const v8bf*)&Bs[buf][rn * APAD + 16 + g * 8];
    }
#pragma unroll
    for (int im = 0; im < 2; ++im)
#pragma unroll
      for (int jn = 0; jn < 4; ++jn)
        acc[im][jn] = WMMA_BF16(a[im].v, b[jn].v, acc[im][jn]);

    async_wait0();
    __syncthreads();
  }

  // epilogue: bias add, split into q/k/v head layouts (v transposed to d-major)
#pragma unroll
  for (int jn = 0; jn < 4; ++jn) {
    const int N = n0 + wn + jn * 16 + ln;
    const float bv = bias[N];
    const int which = N / EE;
    const int cw = N % EE;
    const int h = cw >> 6, d = cw & 63;
#pragma unroll
    for (int im = 0; im < 2; ++im) {
#pragma unroll
      for (int r = 0; r < 8; ++r) {
        const int M = m0 + wm + im * 16 + r + 8 * g;   // C-layout row
        const int bidx = M >> 12, s = M & 4095;
        const int bh = bidx * HH + h;
        const __bf16 hv = (__bf16)(acc[im][jn][r] + bv);
        if (which == 0)      qws[((size_t)bh * SS + s) * HD + d] = hv;
        else if (which == 1) kws[((size_t)bh * SS + s) * HD + d] = hv;
        else                 vws[((size_t)bh * HD + d) * SS + s] = hv;
      }
    }
  }
}

// ---------------- windowed local attention (flash-style) ----------------
__global__ __launch_bounds__(256) void local_attn(
    const __bf16* __restrict__ qws, const __bf16* __restrict__ kws,
    const __bf16* __restrict__ vws, float* __restrict__ out)
{
  __shared__ __bf16 Pb[8][16][136];   // per-wave P staging, 272B stride (bank-clean)
  const int w  = blockIdx.x;          // window
  const int bh = blockIdx.y;          // b*H + h
  const int tid = threadIdx.x;
  const int wv = tid >> 5, lane = tid & 31;
  const int ln = lane & 15, g = lane >> 4;

  // Q fragments for this wave's 16 query rows (K = HD = 64 -> 2 frags)
  Frag qf[2];
  {
    const int srow = w * WSZ + wv * 16 + ln;
    const __bf16* qp = qws + ((size_t)bh * SS + srow) * HD;
#pragma unroll
    for (int f = 0; f < 2; ++f) {
      qf[f].h[0] = *(const v8bf*)(qp + f * 32 + g * 8);
      qf[f].h[1] = *(const v8bf*)(qp + f * 32 + 16 + g * 8);
    }
  }

  const v8f vzero = {0.f,0.f,0.f,0.f,0.f,0.f,0.f,0.f};
  v8f O[4];
  float mrun[8], lrun[8];
#pragma unroll
  for (int dt = 0; dt < 4; ++dt) O[dt] = vzero;
#pragma unroll
  for (int r = 0; r < 8; ++r) { mrun[r] = -1e30f; lrun[r] = 0.f; }

  for (int c = 0; c < 3; ++c) {
    const int wc = w - 1 + c;
    const bool pad = (wc < 0) || (wc >= NWIN);
    const int p0 = wc * WSZ;

    // ---- S = Q K^T (8 key tiles of 16) ----
    v8f sc[8];
#pragma unroll
    for (int t = 0; t < 8; ++t) {
      v8f a = vzero;
#pragma unroll
      for (int f = 0; f < 2; ++f) {
        Frag kb;
        if (pad) {
#pragma unroll
          for (int i = 0; i < 16; ++i) kb.v[i] = (__bf16)(-1.0f);  // reference pad semantics
        } else {
          const int key = p0 + t * 16 + ln;
          const __bf16* kp = kws + ((size_t)bh * SS + key) * HD + f * 32;
          kb.h[0] = *(const v8bf*)(kp + g * 8);
          kb.h[1] = *(const v8bf*)(kp + 16 + g * 8);
        }
        a = WMMA_BF16(qf[f].v, kb.v, a);
      }
      if (!pad) {  // distance mask; pads are NOT masked in the reference
        const int key = p0 + t * 16 + ln;
        const int qb = w * WSZ + wv * 16;
#pragma unroll
        for (int r = 0; r < 8; ++r) {
          const int qi = qb + r + 8 * g;
          if (key > qi + WSZ || key < qi - WSZ) a[r] = -1e30f;
        }
      }
      sc[t] = a;
    }

    // ---- chunk row max (reduce over 8 tiles, then across 16-lane group) ----
    float cm[8];
#pragma unroll
    for (int r = 0; r < 8; ++r) {
      float m = sc[0][r];
#pragma unroll
      for (int t = 1; t < 8; ++t) m = fmaxf(m, sc[t][r]);
      cm[r] = m;
    }
#pragma unroll
    for (int msk = 1; msk < 16; msk <<= 1)
#pragma unroll
      for (int r = 0; r < 8; ++r)
        cm[r] = fmaxf(cm[r], __shfl_xor(cm[r], msk, 32));

    float mnew[8], alpha[8];
#pragma unroll
    for (int r = 0; r < 8; ++r) {
      mnew[r]  = fmaxf(mrun[r], cm[r]);
      alpha[r] = __expf(mrun[r] - mnew[r]);
      mrun[r]  = mnew[r];
      lrun[r] *= alpha[r];
    }
#pragma unroll
    for (int dt = 0; dt < 4; ++dt)
#pragma unroll
      for (int r = 0; r < 8; ++r) O[dt][r] *= alpha[r];

    // ---- P = exp(S - m); row sums; stage P (bf16) into per-wave LDS ----
    float rs[8];
#pragma unroll
    for (int r = 0; r < 8; ++r) rs[r] = 0.f;
#pragma unroll
    for (int t = 0; t < 8; ++t)
#pragma unroll
      for (int r = 0; r < 8; ++r) {
        const float p = __expf(sc[t][r] - mnew[r]);
        sc[t][r] = p;
        rs[r] += p;
      }
#pragma unroll
    for (int msk = 1; msk < 16; msk <<= 1)
#pragma unroll
      for (int r = 0; r < 8; ++r) rs[r] += __shfl_xor(rs[r], msk, 32);
#pragma unroll
    for (int r = 0; r < 8; ++r) lrun[r] += rs[r];

#pragma unroll
    for (int t = 0; t < 8; ++t)
#pragma unroll
      for (int r = 0; r < 8; ++r)
        Pb[wv][r + 8 * g][t * 16 + ln] = (__bf16)sc[t][r];
    // DS ops are in-order within a wave: staging reads below need no wait/barrier

    // ---- O += P V (4 K-frags of 32 keys x 4 d-tiles) ----
#pragma unroll
    for (int kf = 0; kf < 4; ++kf) {
      Frag pf;
      pf.h[0] = *(const v8bf*)&Pb[wv][ln][kf * 32 + g * 8];
      pf.h[1] = *(const v8bf*)&Pb[wv][ln][kf * 32 + 16 + g * 8];
#pragma unroll
      for (int dt = 0; dt < 4; ++dt) {
        Frag vb;
        if (pad) {
#pragma unroll
          for (int i = 0; i < 16; ++i) vb.v[i] = (__bf16)(-1.0f);
        } else {
          const int d = dt * 16 + ln;
          const __bf16* vp = vws + ((size_t)bh * HD + d) * SS + p0 + kf * 32;
          vb.h[0] = *(const v8bf*)(vp + g * 8);
          vb.h[1] = *(const v8bf*)(vp + 16 + g * 8);
        }
        O[dt] = WMMA_BF16(pf.v, vb.v, O[dt]);
      }
    }
  }

  // ---- normalize and write out (B, S, E) fp32 ----
  const int b = bh >> 4, h = bh & 15;
#pragma unroll
  for (int dt = 0; dt < 4; ++dt) {
    const int d = dt * 16 + ln;
#pragma unroll
    for (int r = 0; r < 8; ++r) {
      const int s = w * WSZ + wv * 16 + r + 8 * g;
      out[((size_t)b * SS + s) * EE + h * HD + d] = O[dt][r] / lrun[r];
    }
  }
}

// ---------------- host launcher ----------------
extern "C" void kernel_launch(void* const* d_in, const int* in_sizes, int n_in,
                              void* d_out, int out_size, void* d_ws, size_t ws_size,
                              hipStream_t stream) {
  const float* x  = (const float*)d_in[0];
  const float* wq = (const float*)d_in[1];
  const float* bq = (const float*)d_in[2];
  float* out = (float*)d_out;

  char* ws = (char*)d_ws;
  __bf16* xb  = (__bf16*)(ws);
  __bf16* wT  = (__bf16*)(ws + 33554432);
  __bf16* qws = (__bf16*)(ws + 39845888);
  __bf16* kws = (__bf16*)(ws + 73400320);
  __bf16* vws = (__bf16*)(ws + 106954752);

  cvt_x_kernel <<<16384, 256, 0, stream>>>(x, xb);                 // 16.7M f32 -> bf16
  cvt_wT_kernel<<<12288, 256, 0, stream>>>(wq, wT);                // transpose+convert
  qkv_gemm     <<<dim3(N3E / 128, (BB * SS) / 128), 256, 0, stream>>>(xb, wT, bq, qws, kws, vws);
  local_attn   <<<dim3(NWIN, BHN), 256, 0, stream>>>(qws, kws, vws, out);

  (void)in_sizes; (void)n_in; (void)out_size; (void)ws_size;
}
